// RecGCN_65584150609968
// MI455X (gfx1250) — matrix operand order repaired and checked by reading, not compiled, not run on hardware
//
#include <hip/hip_runtime.h>
#include <hip/hip_bf16.h>
#include <math.h>

// RecGCN for MI455X (gfx1250): B=16, N=1024, D=256, H=128, L=2.
// A_in@x = diag(g)(dep@x) + diag(1-g)(lat@x); A_out uses dep^T/lat^T.
// bf16 WMMA everywhere; adjacency pre-converted to bf16 (fits 192MB L2);
// main GEMM uses a double-buffered global_load_async_to_lds_b128 pipeline.

#define BB 16
#define NN 1024
#define DD 256
#define HH 128
#define LL 2

typedef __bf16 bf16;
typedef __attribute__((ext_vector_type(16))) __bf16 v16bf;
typedef __attribute__((ext_vector_type(8)))  __bf16 v8bf;
typedef __attribute__((ext_vector_type(8)))  float  v8f;
typedef __attribute__((ext_vector_type(4)))  float  v4f;

__device__ __forceinline__ v8f wmma_bf16(v16bf a, v16bf b, v8f c) {
  return __builtin_amdgcn_wmma_f32_16x16x32_bf16(false, a, false, b, (short)0, c,
                                                 false, false);
}

__device__ __forceinline__ v8f zero8() {
  v8f z = {0.f, 0.f, 0.f, 0.f, 0.f, 0.f, 0.f, 0.f};
  return z;
}

__device__ __forceinline__ float wave_sum(float v) {
#pragma unroll
  for (int off = 16; off > 0; off >>= 1) v += __shfl_xor(v, off, 32);
  return v;
}

__device__ __forceinline__ v8bf cvt8(v4f a, v4f b) {
  v8bf o;
#pragma unroll
  for (int j = 0; j < 4; ++j) { o[j] = (bf16)a[j]; o[4 + j] = (bf16)b[j]; }
  return o;
}

// A fragment from m-major LDS tile [M][ld]: lane m, elements = two contiguous
// 8-bf16 runs at K = kh and K = 16+kh  (ISA 7.12.2 16-bit A layout).
__device__ __forceinline__ v16bf frag_a_mm(const bf16* As, int ld, int r, int lane) {
  int m  = r + (lane & 15);
  int kh = (lane >> 4) << 3;  // 0 or 8
  v8bf lo = *(const v8bf*)(As + (size_t)m * ld + kh);
  v8bf hi = *(const v8bf*)(As + (size_t)m * ld + 16 + kh);
  return __builtin_shufflevector(lo, hi, 0, 1, 2, 3, 4, 5, 6, 7,
                                 8, 9, 10, 11, 12, 13, 14, 15);
}

// B fragment from n-major LDS tile [N][ld]: lane n, elements = two contiguous
// 8-bf16 runs at K = ko and K = 16+ko (matches A layout).
__device__ __forceinline__ v16bf frag_b_nm(const bf16* Bs, int ld, int c, int lane) {
  int n  = c + (lane & 15);
  int ko = (lane >> 4) << 3;  // 0 or 8
  v8bf lo = *(const v8bf*)(Bs + (size_t)n * ld + ko);
  v8bf hi = *(const v8bf*)(Bs + (size_t)n * ld + ko + 16);
  return __builtin_shufflevector(lo, hi, 0, 1, 2, 3, 4, 5, 6, 7,
                                 8, 9, 10, 11, 12, 13, 14, 15);
}

// --- Kernel 0: adjacency f32 -> bf16 (runs once per call; lives in L2) -----
__global__ void k_adj_cvt(const float* __restrict__ dep, const float* __restrict__ lat,
                          bf16* __restrict__ depb, bf16* __restrict__ latb) {
  const float* src = blockIdx.y ? lat : dep;
  bf16* dst = blockIdx.y ? latb : depb;
  size_t i = ((size_t)blockIdx.x * 256 + threadIdx.x) * 8;
  v4f a = *(const v4f*)(src + i);
  v4f b = *(const v4f*)(src + i + 4);
  *(v8bf*)(dst + i) = cvt8(a, b);
}

// --- Kernel 1: refine gate + bf16 copy of x --------------------------------
__global__ void k_gate_cvt(const float* __restrict__ x, const float* __restrict__ rg,
                           bf16* __restrict__ xb, float* __restrict__ gate) {
  int tid = threadIdx.x, lane = tid & 31;
  int row = blockIdx.x * 8 + (tid >> 5);
  const float* xr = x + (size_t)row * DD;
  bf16* xbr = xb + (size_t)row * DD;
  float acc = 0.f;
#pragma unroll
  for (int j = 0; j < 8; ++j) {
    int d = lane + j * 32;
    float v = xr[d];
    acc += v * rg[d];
    xbr[d] = (bf16)v;
  }
  acc = wave_sum(acc);
  if (lane == 0) gate[row] = 1.f / (1.f + __expf(-acc));
}

// --- Kernel 2: blended aggregation GEMM, double-buffered async pipeline ----
template <bool TRANSA>
__global__ void __launch_bounds__(256) k_gemm_agg(
    const bf16* __restrict__ depb, const bf16* __restrict__ latb,
    const bf16* __restrict__ xb, const float* __restrict__ gate,
    float* __restrict__ out) {
  __shared__ bf16 Ad[2][64][40];   // m-major [M][K], 80B rows
  __shared__ bf16 Al[2][64][40];
  __shared__ bf16 Xs[2][128][40];  // n-major [D][K]
  int tid = threadIdx.x, lane = tid & 31, w = tid >> 5;
  int i0 = blockIdx.x * 64, d0 = blockIdx.y * 128, b = blockIdx.z;
  int r = (w & 3) * 16, c = (w >> 2) * 64;
  const size_t adjBase = (size_t)b * NN * NN;
  const size_t xBase   = (size_t)b * NN * DD;

  v8f accd[4], accl[4];
#pragma unroll
  for (int t = 0; t < 4; ++t) { accd[t] = zero8(); accl[t] = zero8(); }

  auto stage = [&](int kt, int buf) {
    int k0 = kt * 32;
    if (!TRANSA) {
      // A tiles: pure bf16 copy, 16B granular -> async DMA straight into LDS.
      int rr = tid >> 2;            // 0..63 (output row in tile)
      int kq = (tid & 3) * 8;       // 0,8,16,24
      const bf16* pd = depb + adjBase + (size_t)(i0 + rr) * NN + k0 + kq;
      const bf16* pl = latb + adjBase + (size_t)(i0 + rr) * NN + k0 + kq;
      unsigned od = (unsigned)(uintptr_t)&Ad[buf][rr][kq];
      unsigned ol = (unsigned)(uintptr_t)&Al[buf][rr][kq];
      asm volatile("global_load_async_to_lds_b128 %0, %1, off"
                   :: "v"(od), "v"(pd) : "memory");
      asm volatile("global_load_async_to_lds_b128 %0, %1, off"
                   :: "v"(ol), "v"(pl) : "memory");
    } else {
      // A^T tiles: row-chunks of dep[k, i..] transposed into m-major LDS.
      int kk = tid >> 3;            // 0..31 (K row)
      int mq = (tid & 7) * 8;       // 0..56 (output row in tile)
      const bf16* pd = depb + adjBase + (size_t)(k0 + kk) * NN + i0 + mq;
      const bf16* pl = latb + adjBase + (size_t)(k0 + kk) * NN + i0 + mq;
      v8bf dv = *(const v8bf*)pd;
      v8bf lv = *(const v8bf*)pl;
#pragma unroll
      for (int j = 0; j < 8; ++j) {
        Ad[buf][mq + j][kk] = dv[j];
        Al[buf][mq + j][kk] = lv[j];
      }
    }
    {
      // X tile: vector global loads, transpose into n-major LDS.
      int kk = tid >> 3;            // 0..31
      int dq = (tid & 7) * 16;      // 0..112
      const bf16* px = xb + xBase + (size_t)(k0 + kk) * DD + d0 + dq;
      v8bf x0 = *(const v8bf*)px;
      v8bf x1 = *(const v8bf*)(px + 8);
#pragma unroll
      for (int j = 0; j < 8; ++j) {
        Xs[buf][dq + j][kk]     = x0[j];
        Xs[buf][dq + 8 + j][kk] = x1[j];
      }
    }
  };

  stage(0, 0);
  for (int kt = 0; kt < NN / 32; ++kt) {
    int cur = kt & 1;
    // Wait for this buffer's async transfers (issued last iteration), then
    // rendezvous; next tile's transfers stay in flight through the WMMAs.
    if (!TRANSA) asm volatile("s_wait_asynccnt 0x0" ::: "memory");
    __syncthreads();
    if (kt + 1 < NN / 32) stage(kt + 1, cur ^ 1);

    v16bf ad = frag_a_mm(&Ad[cur][0][0], 40, r, lane);
    v16bf al = frag_a_mm(&Al[cur][0][0], 40, r, lane);
#pragma unroll
    for (int t = 0; t < 4; ++t) {
      v16bf bx = frag_b_nm(&Xs[cur][0][0], 40, c + t * 16, lane);
      accd[t] = wmma_bf16(ad, bx, accd[t]);
      accl[t] = wmma_bf16(al, bx, accl[t]);
    }
  }

  int n2 = lane & 15, hi = lane >> 4;
#pragma unroll
  for (int rr = 0; rr < 8; ++rr) {
    int gr = i0 + r + rr + hi * 8;   // C layout: VGPR rr -> M=rr (+8 hi lanes)
    float g = gate[b * NN + gr];
    float* po = out + ((size_t)b * NN + gr) * DD + d0 + c + n2;
#pragma unroll
    for (int t = 0; t < 4; ++t)
      po[t * 16] = g * accd[t][rr] + (1.f - g) * accl[t][rr];
  }
}

// --- Kernel 3: GCN scalar gate + residual add, emit bf16 -------------------
__global__ void k_gcn_gate(const float* __restrict__ ax, const float* __restrict__ x,
                           const float* __restrict__ Wg, const float* __restrict__ b0,
                           bf16* __restrict__ sb) {
  int tid = threadIdx.x, lane = tid & 31;
  int row = blockIdx.x * 8 + (tid >> 5);
  const float* ar = ax + (size_t)row * DD;
  const float* xr = x + (size_t)row * DD;
  bf16* sr = sb + (size_t)row * DD;
  float acc = 0.f, av[8];
#pragma unroll
  for (int j = 0; j < 8; ++j) {
    int d = lane + j * 32;
    av[j] = ar[d];
    acc += av[j] * Wg[d];
  }
  acc = wave_sum(acc);
  float sg = 1.f / (1.f + __expf(-acc));
  float bb = b0[0];
#pragma unroll
  for (int j = 0; j < 8; ++j) {
    int d = lane + j * 32;
    sr[d] = (bf16)(av[j] * sg + bb + xr[d]);
  }
}

// --- Kernel 4: FC GEMM  h = sb @ W^T + 2b ----------------------------------
__global__ void __launch_bounds__(256) k_fc(const bf16* __restrict__ sb,
                                            const float* __restrict__ W,
                                            const float* __restrict__ bias,
                                            float* __restrict__ hbuf, int colOff) {
  __shared__ bf16 Ws[128][40];  // n-major [H][K]
  __shared__ bf16 Ss[64][40];   // m-major [M][K]
  int tid = threadIdx.x, lane = tid & 31, w = tid >> 5;
  int r0 = blockIdx.x * 64;
  int r = (w & 3) * 16, c = (w >> 2) * 64;
  v8f acc[4];
#pragma unroll
  for (int t = 0; t < 4; ++t) acc[t] = zero8();

  for (int kt = 0; kt < DD / 32; ++kt) {
    int k0 = kt * 32;
    {
      int h = tid >> 1;             // 0..127
      int kq = (tid & 1) * 16;      // 0 or 16
      const float* pw = W + (size_t)h * DD + k0 + kq;
      v4f w0 = *(const v4f*)pw,       w1 = *(const v4f*)(pw + 4);
      v4f w2 = *(const v4f*)(pw + 8), w3 = *(const v4f*)(pw + 12);
      *(v8bf*)&Ws[h][kq]     = cvt8(w0, w1);
      *(v8bf*)&Ws[h][kq + 8] = cvt8(w2, w3);
    }
    {
      int m = tid >> 2;             // 0..63
      int kq = (tid & 3) * 8;       // 0..24
      const bf16* ps = sb + (size_t)(r0 + m) * DD + k0 + kq;
      *(v8bf*)&Ss[m][kq] = *(const v8bf*)ps;
    }
    __syncthreads();
    v16bf a = frag_a_mm(&Ss[0][0], 40, r, lane);
#pragma unroll
    for (int t = 0; t < 4; ++t)
      acc[t] = wmma_bf16(a, frag_b_nm(&Ws[0][0], 40, c + t * 16, lane), acc[t]);
    __syncthreads();
  }

  int n2 = lane & 15, hi = lane >> 4;
#pragma unroll
  for (int rr = 0; rr < 8; ++rr) {
    int gr = r0 + r + rr + hi * 8;
    float* po = hbuf + (size_t)gr * (2 * HH) + colOff + c + n2;
#pragma unroll
    for (int t = 0; t < 4; ++t)
      po[t * 16] = acc[t][rr] + 2.f * bias[c + t * 16 + n2];
  }
}

// --- Kernel 5: LayerNorm + exact GELU --------------------------------------
__global__ void k_ln_gelu(const float* __restrict__ h, const float* __restrict__ g,
                          const float* __restrict__ bt, float* __restrict__ out) {
  int tid = threadIdx.x, lane = tid & 31;
  int row = blockIdx.x * 8 + (tid >> 5);
  const float* hr = h + (size_t)row * (2 * HH);
  float* orow = out + (size_t)row * (2 * HH);
  float v[8], s = 0.f, sq = 0.f;
#pragma unroll
  for (int j = 0; j < 8; ++j) {
    v[j] = hr[lane + j * 32];
    s += v[j];
    sq += v[j] * v[j];
  }
  s = wave_sum(s);
  sq = wave_sum(sq);
  float mu = s * (1.f / 256.f);
  float var = sq * (1.f / 256.f) - mu * mu;
  float rstd = rsqrtf(var + 1e-5f);
#pragma unroll
  for (int j = 0; j < 8; ++j) {
    int d = lane + j * 32;
    float y = (v[j] - mu) * rstd * g[d] + bt[d];
    orow[d] = 0.5f * y * (1.f + erff(y * 0.70710678118654752f));
  }
}

// --- Host launch ------------------------------------------------------------
extern "C" void kernel_launch(void* const* d_in, const int* in_sizes, int n_in,
                              void* d_out, int out_size, void* d_ws, size_t ws_size,
                              hipStream_t stream) {
  (void)in_sizes; (void)n_in; (void)out_size; (void)ws_size;
  const float* x0  = (const float*)d_in[0];
  const float* lat = (const float*)d_in[1];
  const float* dep = (const float*)d_in[2];
  const float* rg  = (const float*)d_in[3];
  const float* Wgi = (const float*)d_in[4];
  const float* bgi = (const float*)d_in[5];
  const float* Wgo = (const float*)d_in[6];
  const float* bgo = (const float*)d_in[7];
  const float* Wi  = (const float*)d_in[8];
  const float* bi  = (const float*)d_in[9];
  const float* Wo  = (const float*)d_in[10];
  const float* bo  = (const float*)d_in[11];
  const float* lng = (const float*)d_in[12];
  const float* lnb = (const float*)d_in[13];
  float* outp = (float*)d_out;

  char* ws = (char*)d_ws;
  size_t off = 0;
  auto alloc = [&](size_t bytes) -> void* {
    off = (off + 255) & ~(size_t)255;
    void* p = ws + off;
    off += bytes;
    return p;
  };
  const size_t BN = (size_t)BB * NN;
  bf16*  depb  = (bf16*)alloc((size_t)BB * NN * NN * 2);
  bf16*  latb  = (bf16*)alloc((size_t)BB * NN * NN * 2);
  bf16*  xb    = (bf16*)alloc(BN * DD * 2);
  float* gate  = (float*)alloc(BN * 4);
  float* axin  = (float*)alloc(BN * DD * 4);
  float* axout = (float*)alloc(BN * DD * 4);
  bf16*  sbin  = (bf16*)alloc(BN * DD * 2);
  bf16*  sbout = (bf16*)alloc(BN * DD * 2);
  float* hbuf  = (float*)alloc(BN * 2 * HH * 4);
  float* xcur  = (float*)alloc(BN * DD * 4);

  dim3 rowGrid((unsigned)(BN / 8));
  dim3 gemmGrid(NN / 64, DD / 128, BB);
  dim3 fcGrid((unsigned)(BN / 64));
  dim3 cvtGrid((unsigned)((size_t)BB * NN * NN / 8 / 256), 2);

  k_adj_cvt<<<cvtGrid, 256, 0, stream>>>(dep, lat, depb, latb);

  for (int l = 0; l < LL; ++l) {
    const float* xin = (l == 0) ? x0 : xcur;
    float* hdst = (l == LL - 1) ? outp : xcur;
    k_gate_cvt<<<rowGrid, 256, 0, stream>>>(xin, rg + l * DD, xb, gate);
    k_gemm_agg<false><<<gemmGrid, 256, 0, stream>>>(depb, latb, xb, gate, axin);
    k_gemm_agg<true ><<<gemmGrid, 256, 0, stream>>>(depb, latb, xb, gate, axout);
    k_gcn_gate<<<rowGrid, 256, 0, stream>>>(axin,  xin, Wgi + l * DD, bgi + l, sbin);
    k_gcn_gate<<<rowGrid, 256, 0, stream>>>(axout, xin, Wgo + l * DD, bgo + l, sbout);
    k_fc<<<fcGrid, 256, 0, stream>>>(sbin,  Wi + (size_t)l * HH * DD, bi + l * HH, hbuf, 0);
    k_fc<<<fcGrid, 256, 0, stream>>>(sbout, Wo + (size_t)l * HH * DD, bo + l * HH, hbuf, HH);
    k_ln_gelu<<<rowGrid, 256, 0, stream>>>(hbuf, lng + l * 2 * HH, lnb + l * 2 * HH, hdst);
  }
}
